// MFN_56831007260961
// MI455X (gfx1250) — matrix operand order repaired and checked by reading, not compiled, not run on hardware
//
#include <hip/hip_runtime.h>
#include <hip/hip_bf16.h>

// ---------------------------------------------------------------------------
// MFN (memory fusion network) on gfx1250.
// Strategy: batch rows are independent across the whole sequence -> one
// persistent kernel, 64 workgroups x 16 batch rows, t-loop inside the kernel.
// All GEMMs use v_wmma_f32_16x16x32_bf16 (fp32 accum). Weights are converted
// once per launch into bf16 zero-padded [Npad][Kpad] row-major in d_ws, which
// makes the WMMA B-fragment a single contiguous 32B global load per lane.
// Recurrent state (c, mem) stays fp32 in LDS across all 128 steps.
// ---------------------------------------------------------------------------

#define T_STEPS 128
#define NBATCH  1024
#define DIN     300
#define DHID    100

typedef __attribute__((ext_vector_type(16))) __bf16 v16bf;
typedef __attribute__((ext_vector_type(8)))  __bf16 v8bf;
typedef __attribute__((ext_vector_type(8)))  float  v8f;

__device__ __forceinline__ __bf16 f2bf(float f) {
  unsigned u = __builtin_bit_cast(unsigned, f);
  unsigned r = (u + 0x7FFFu + ((u >> 16) & 1u)) >> 16;   // round-to-nearest-even
  return __builtin_bit_cast(__bf16, (unsigned short)r);
}
__device__ __forceinline__ float bf2f(__bf16 b) {
  unsigned u = ((unsigned)__builtin_bit_cast(unsigned short, b)) << 16;
  return __builtin_bit_cast(float, u);
}
__device__ __forceinline__ float sigmoidf(float x) { return 1.f / (1.f + __expf(-x)); }

// ---------------- workspace layout (bf16 elements) -------------------------
constexpr size_t SZ_WIH   = 400 * 320;       // Wih: 400x300 -> 400x320
constexpr size_t SZ_WHH   = 400 * 128;       // Whh: 400x100 -> 400x128
constexpr size_t OFF_WIH  = 0;
constexpr size_t OFF_WHH  = OFF_WIH  + 3 * SZ_WIH;
constexpr size_t OFF_A1W1 = OFF_WHH  + 3 * SZ_WHH;   // 100x600 -> 112x608
constexpr size_t OFF_A1W2 = OFF_A1W1 + 112 * 608;    // 600x100 -> 608x128
constexpr size_t OFF_A2W1 = OFF_A1W2 + 608 * 128;    // 100x600 -> 112x608
constexpr size_t OFF_A2W2 = OFF_A2W1 + 112 * 608;    // 100x100 -> 112x128
constexpr size_t OFF_G1W1 = OFF_A2W2 + 112 * 128;    // 100x700 -> 112x736 (gap @600)
constexpr size_t OFF_G1W2 = OFF_G1W1 + 112 * 736;    // 100x100 -> 112x128
constexpr size_t OFF_G2W1 = OFF_G1W2 + 112 * 128;
constexpr size_t OFF_G2W2 = OFF_G2W1 + 112 * 736;
constexpr size_t WS_ELEMS = OFF_G2W2 + 112 * 128;    // ~0.96M elems = ~1.9 MB

// ------------- weight prep: fp32 [outN][inN] -> bf16 [outPad][inPad] --------
// Optional column gap: dst cols [gapStart, gapStart+gapLen) are zero and the
// remaining source columns shift right (used for [attended|pad8|mem] layout).
__global__ void prep_weight(const float* __restrict__ src, __bf16* __restrict__ dst,
                            int outN, int inN, int outPad, int inPad,
                            int gapStart, int gapLen) {
  int idx = blockIdx.x * 256 + threadIdx.x;
  if (idx >= outPad * inPad) return;
  int r = idx / inPad, c = idx % inPad;
  float v = 0.f;
  if (r < outN) {
    if (c < gapStart) {
      if (c < inN) v = src[(size_t)r * inN + c];
    } else if (c >= gapStart + gapLen) {
      int sc = c - gapLen;
      if (sc < inN) v = src[(size_t)r * inN + sc];
    }
  }
  dst[idx] = f2bf(v);
}

// ------------- wave-level GEMM: C = act(A1@W1^T [+ A2@W2^T] + bias) ---------
// A: bf16 LDS, 16 rows, row stride sA elements, Kpad = ks*32 (zero padded).
// W: bf16 global [Npad][sW] row-major (pre-padded).  One 16x16 output tile
// per iteration; tiles strided across the 8 waves of the block.
// ACT: 0 none, 1 relu, 2 tanh, 3 sigmoid.  OUTBF: write bf16 else f32.
template <int ACT, bool OUTBF>
__device__ __forceinline__ void wave_gemm(
    const __bf16* __restrict__ A1, int sA1, int ks1,
    const __bf16* __restrict__ W1, int sW1,
    const __bf16* __restrict__ A2, int sA2, int ks2,
    const __bf16* __restrict__ W2, int sW2,
    int nTiles, const float* __restrict__ bias1, const float* __restrict__ bias2,
    int realN, void* __restrict__ C, int sC, int lane, int wave, int nWaves) {
  const int row = lane & 15;   // A row / D column index
  const int hi  = lane >> 4;   // K-half select
  for (int t = wave; t < nTiles; t += nWaves) {
    v8f acc = {};
    const int col = t * 16 + row;
    {
      const __bf16* a = A1 + row * sA1 + hi * 16;
      const __bf16* w = W1 + (size_t)col * sW1 + hi * 16;
      for (int k = 0; k < ks1; ++k) {
        union { v16bf v; v8bf h[2]; } af;
        af.h[0] = *(const v8bf*)(a + k * 32);        // K 0-7  (or 16-23)
        af.h[1] = *(const v8bf*)(a + k * 32 + 8);    // K 8-15 (or 24-31)
        v16bf bf = *(const v16bf*)(w + k * 32);      // 16 contiguous K
        acc = __builtin_amdgcn_wmma_f32_16x16x32_bf16(
            false, af.v, false, bf, (short)0, acc, false, false);
      }
    }
    if (ks2 > 0) {
      const __bf16* a = A2 + row * sA2 + hi * 16;
      const __bf16* w = W2 + (size_t)col * sW2 + hi * 16;
      for (int k = 0; k < ks2; ++k) {
        union { v16bf v; v8bf h[2]; } af;
        af.h[0] = *(const v8bf*)(a + k * 32);
        af.h[1] = *(const v8bf*)(a + k * 32 + 8);
        v16bf bf = *(const v16bf*)(w + k * 32);
        acc = __builtin_amdgcn_wmma_f32_16x16x32_bf16(
            false, af.v, false, bf, (short)0, acc, false, false);
      }
    }
    float bsum = 0.f;
    if (col < realN) {
      if (bias1) bsum += bias1[col];
      if (bias2) bsum += bias2[col];
    }
#pragma unroll
    for (int r = 0; r < 8; ++r) {
      float v = acc[r] + bsum;
      if (ACT == 1) v = fmaxf(v, 0.f);
      if (ACT == 2) v = tanhf(v);
      if (ACT == 3) v = sigmoidf(v);
      int m = hi * 8 + r;                            // D row
      if (OUTBF) ((__bf16*)C)[m * sC + col] = f2bf(v);
      else       ((float*) C)[m * sC + col] = v;
    }
  }
}

struct MfnParams {
  const float* x;
  float* out;
  const __bf16* ws;
  const float* bih[3]; const float* bhh[3];
  const float* att1_b1; const float* att1_b2;
  const float* att2_b1; const float* att2_b2;
  const float* g1_b1;   const float* g1_b2;
  const float* g2_b1;   const float* g2_b2;
};

__global__ __launch_bounds__(256) void mfn_persistent(MfnParams p) {
  // ---- LDS (~132 KB; CDNA5 allows up to 320 KB per workgroup) ----
  __shared__ __align__(16) __bf16 s_x[16 * 320];        // x tile, one modality
  __shared__ __align__(16) __bf16 s_h[3][16 * 128];     // h (bf16, padded)
  __shared__              float   s_c[3][16 * 100];     // c (fp32 state)
  __shared__ __align__(16) float  s_f32[16 * 608];      // gates / logits / chat+g1+g2
  __shared__ __align__(16) __bf16 s_cstar[16 * 608];    // [c_prev | c_new]
  __shared__ __align__(16) __bf16 s_both[16 * 736];     // [attended | gap8 | mem | pad]
  __shared__ __align__(16) __bf16 s_hid[16 * 128];      // MLP hidden
  __shared__              float   s_mem[16 * 100];      // mem (fp32 state)
  __shared__              float   s_red[16 * 16];       // softmax reduction

  const int tid   = threadIdx.x;
  const int lane  = tid & 31;     // wave32
  const int wave  = tid >> 5;
  const int NW    = 8;
  const int nbase = blockIdx.x * 16;

  // zero all state + pad regions once
  {
    __bf16 z = __builtin_bit_cast(__bf16, (unsigned short)0);
    for (int i = tid; i < 16 * 320; i += 256) s_x[i] = z;
    for (int i = tid; i < 3 * 16 * 128; i += 256) ((__bf16*)s_h)[i] = z;
    for (int i = tid; i < 3 * 16 * 100; i += 256) ((float*)s_c)[i] = 0.f;
    for (int i = tid; i < 16 * 608; i += 256) { s_f32[i] = 0.f; s_cstar[i] = z; }
    for (int i = tid; i < 16 * 736; i += 256) s_both[i] = z;
    for (int i = tid; i < 16 * 128; i += 256) s_hid[i] = z;
    for (int i = tid; i < 16 * 100; i += 256) s_mem[i] = 0.f;
    for (int i = tid; i < 16 * 16;  i += 256) s_red[i] = 0.f;
  }
  __syncthreads();

  const __bf16* ws = p.ws;

  for (int t = 0; t < T_STEPS; ++t) {
    // 1) cStar[:, 0:300] = previous c (before LSTM update)
    for (int i = tid; i < 16 * 300; i += 256) {
      int r = i / 300, j = i % 300;
      s_cstar[r * 608 + j] = f2bf(s_c[j / 100][r * 100 + j % 100]);
    }
    __syncthreads();

    // 2) three LSTM cells
    for (int m = 0; m < 3; ++m) {
      const float* xsrc = p.x + ((size_t)t * NBATCH + nbase) * (3 * DIN) + m * DIN;
      for (int i = tid; i < 16 * 300; i += 256) {
        int r = i / 300, j = i % 300;
        s_x[r * 320 + j] = f2bf(xsrc[(size_t)r * (3 * DIN) + j]);
      }
      __syncthreads();
      // gates = x@Wih^T + h@Whh^T + bih + bhh  -> s_f32 cols 0..399
      wave_gemm<0, false>(s_x, 320, 10, ws + OFF_WIH + m * SZ_WIH, 320,
                          s_h[m], 128, 4, ws + OFF_WHH + m * SZ_WHH, 128,
                          25, p.bih[m], p.bhh[m], 400, s_f32, 608, lane, wave, NW);
      __syncthreads();
      for (int i = tid; i < 16 * 100; i += 256) {
        int r = i / 100, n = i % 100;
        float gi = s_f32[r * 608 + n];
        float gf = s_f32[r * 608 + 100 + n];
        float gg = s_f32[r * 608 + 200 + n];
        float go = s_f32[r * 608 + 300 + n];
        float cnew = sigmoidf(gf) * s_c[m][i] + sigmoidf(gi) * tanhf(gg);
        float hnew = sigmoidf(go) * tanhf(cnew);
        s_c[m][i] = cnew;
        s_h[m][r * 128 + n] = f2bf(hnew);
        s_cstar[r * 608 + 300 + m * 100 + n] = f2bf(cnew);
        p.out[((size_t)t * NBATCH + nbase + r) * 400 + m * 100 + n] = hnew;
      }
      __syncthreads();
    }

    // 3) att1 hidden = relu(cStar @ w1^T + b1)  -> s_hid (bf16)
    wave_gemm<1, true>(s_cstar, 608, 19, ws + OFF_A1W1, 608,
                       nullptr, 0, 0, nullptr, 0,
                       7, p.att1_b1, nullptr, 100, s_hid, 128, lane, wave, NW);
    __syncthreads();
    // 4) logits = hidden @ w2^T + b2  -> s_f32 cols 0..599
    wave_gemm<0, false>(s_hid, 128, 4, ws + OFF_A1W2, 128,
                        nullptr, 0, 0, nullptr, 0,
                        38, p.att1_b2, nullptr, 600, s_f32, 608, lane, wave, NW);
    __syncthreads();

    // 5) softmax over feature axis, attended = attn * cStar -> s_both[:,0:600]
    {
      int r = tid >> 4, sub = tid & 15;
      float mx = -1e30f;
      for (int c = sub; c < 600; c += 16) mx = fmaxf(mx, s_f32[r * 608 + c]);
      s_red[r * 16 + sub] = mx;
      __syncthreads();
      float rmx = -1e30f;
      for (int k = 0; k < 16; ++k) rmx = fmaxf(rmx, s_red[r * 16 + k]);
      __syncthreads();
      float ps = 0.f;
      for (int c = sub; c < 600; c += 16) {
        float e = __expf(s_f32[r * 608 + c] - rmx);
        s_f32[r * 608 + c] = e;
        ps += e;
      }
      s_red[r * 16 + sub] = ps;
      __syncthreads();
      float rsum = 0.f;
      for (int k = 0; k < 16; ++k) rsum += s_red[r * 16 + k];
      float rinv = 1.f / rsum;
      for (int c = sub; c < 600; c += 16) {
        float v = s_f32[r * 608 + c] * rinv * bf2f(s_cstar[r * 608 + c]);
        s_both[r * 736 + c] = f2bf(v);
      }
      for (int i = tid; i < 16 * 100; i += 256)   // mem -> s_both[:,608:708]
        s_both[(i / 100) * 736 + 608 + i % 100] = f2bf(s_mem[i]);
    }
    __syncthreads();

    // 6) cHat = tanh(mlp(attended))   (K=608 uses the zero gap cols 600-607)
    wave_gemm<1, true>(s_both, 736, 19, ws + OFF_A2W1, 608,
                       nullptr, 0, 0, nullptr, 0,
                       7, p.att2_b1, nullptr, 100, s_hid, 128, lane, wave, NW);
    __syncthreads();
    wave_gemm<2, false>(s_hid, 128, 4, ws + OFF_A2W2, 128,
                        nullptr, 0, 0, nullptr, 0,
                        7, p.att2_b2, nullptr, 100, s_f32, 608, lane, wave, NW); // chat cols 0-111
    __syncthreads();

    // 7) gamma1 = sigmoid(mlp(both)), K=736
    wave_gemm<1, true>(s_both, 736, 23, ws + OFF_G1W1, 736,
                       nullptr, 0, 0, nullptr, 0,
                       7, p.g1_b1, nullptr, 100, s_hid, 128, lane, wave, NW);
    __syncthreads();
    wave_gemm<3, false>(s_hid, 128, 4, ws + OFF_G1W2, 128,
                        nullptr, 0, 0, nullptr, 0,
                        7, p.g1_b2, nullptr, 100, s_f32 + 112, 608, lane, wave, NW);
    __syncthreads();

    // 8) gamma2
    wave_gemm<1, true>(s_both, 736, 23, ws + OFF_G2W1, 736,
                       nullptr, 0, 0, nullptr, 0,
                       7, p.g2_b1, nullptr, 100, s_hid, 128, lane, wave, NW);
    __syncthreads();
    wave_gemm<3, false>(s_hid, 128, 4, ws + OFF_G2W2, 128,
                        nullptr, 0, 0, nullptr, 0,
                        7, p.g2_b2, nullptr, 100, s_f32 + 224, 608, lane, wave, NW);
    __syncthreads();

    // 9) mem = g1*mem + g2*cHat ; write mem slice of output
    for (int i = tid; i < 16 * 100; i += 256) {
      int r = i / 100, n = i % 100;
      float m2 = s_f32[r * 608 + 112 + n] * s_mem[i] +
                 s_f32[r * 608 + 224 + n] * s_f32[r * 608 + n];
      s_mem[i] = m2;
      p.out[((size_t)t * NBATCH + nbase + r) * 400 + 300 + n] = m2;
    }
    __syncthreads();
  }
}

extern "C" void kernel_launch(void* const* d_in, const int* in_sizes, int n_in,
                              void* d_out, int out_size, void* d_ws, size_t ws_size,
                              hipStream_t stream) {
  (void)in_sizes; (void)n_in; (void)out_size; (void)ws_size;  // needs ~1.9 MB of d_ws
  __bf16* ws = (__bf16*)d_ws;

  auto prep = [&](int inIdx, size_t off, int outN, int inN, int outPad, int inPad,
                  int gapStart, int gapLen) {
    int total = outPad * inPad;
    prep_weight<<<(total + 255) / 256, 256, 0, stream>>>(
        (const float*)d_in[inIdx], ws + off, outN, inN, outPad, inPad, gapStart, gapLen);
  };
  for (int m = 0; m < 3; ++m) {
    prep(1 + m * 4, OFF_WIH + m * SZ_WIH, 400, 300, 400, 320, 320, 0);  // Wih_m
    prep(2 + m * 4, OFF_WHH + m * SZ_WHH, 400, 100, 400, 128, 128, 0);  // Whh_m
  }
  prep(13, OFF_A1W1, 100, 600, 112, 608, 608, 0);
  prep(15, OFF_A1W2, 600, 100, 608, 128, 128, 0);
  prep(17, OFF_A2W1, 100, 600, 112, 608, 608, 0);
  prep(19, OFF_A2W2, 100, 100, 112, 128, 128, 0);
  prep(21, OFF_G1W1, 100, 700, 112, 736, 600, 8);  // gap for [attended|mem]
  prep(23, OFF_G1W2, 100, 100, 112, 128, 128, 0);
  prep(25, OFF_G2W1, 100, 700, 112, 736, 600, 8);
  prep(27, OFF_G2W2, 100, 100, 112, 128, 128, 0);

  MfnParams p;
  p.x = (const float*)d_in[0];
  p.out = (float*)d_out;
  p.ws = ws;
  p.bih[0] = (const float*)d_in[3];  p.bhh[0] = (const float*)d_in[4];
  p.bih[1] = (const float*)d_in[7];  p.bhh[1] = (const float*)d_in[8];
  p.bih[2] = (const float*)d_in[11]; p.bhh[2] = (const float*)d_in[12];
  p.att1_b1 = (const float*)d_in[14]; p.att1_b2 = (const float*)d_in[16];
  p.att2_b1 = (const float*)d_in[18]; p.att2_b2 = (const float*)d_in[20];
  p.g1_b1 = (const float*)d_in[22];   p.g1_b2 = (const float*)d_in[24];
  p.g2_b1 = (const float*)d_in[26];   p.g2_b2 = (const float*)d_in[28];

  mfn_persistent<<<NBATCH / 16, 256, 0, stream>>>(p);
}
